// DynamicQuantizeLinear_63230508531827
// MI455X (gfx1250) — 8-bit matrix operand order repaired and checked by reading, compiled, not run on hardware
//
#include <hip/hip_runtime.h>
#include <stdint.h>


typedef __attribute__((ext_vector_type(8))) int vi8;

#define MAX_I8 127.0f

__device__ __forceinline__ int clamp_q(float v, float inv) {
  int q = (int)rintf(v * inv);
  q = q > 127 ? 127 : q;
  q = q < -127 ? -127 : q;
  return q & 255;
}

__global__ void quant_rows_i8(const float* __restrict__ src,
                              int8_t* __restrict__ dst,
                              float* __restrict__ scales,
                              int rowlen, float floor_val) {
  __shared__ float red[256];
  const int row = blockIdx.x;
  const float4* srow = (const float4*)(src + (size_t)row * rowlen);
  const int nvec = rowlen >> 2;

  float m = 0.0f;
  for (int i = threadIdx.x; i < nvec; i += 256) {
    float4 v = srow[i];
    m = fmaxf(m, fmaxf(fmaxf(fabsf(v.x), fabsf(v.y)),
                       fmaxf(fabsf(v.z), fabsf(v.w))));
  }
  red[threadIdx.x] = m;
  __syncthreads();
  for (int s = 128; s > 0; s >>= 1) {
    if (threadIdx.x < s)
      red[threadIdx.x] = fmaxf(red[threadIdx.x], red[threadIdx.x + s]);
    __syncthreads();
  }

  const float amax = fmaxf(red[0], floor_val);
  const float scale = amax * (1.0f / MAX_I8);
  const float inv = MAX_I8 / amax;
  if (threadIdx.x == 0) scales[row] = scale;

  int* drow = (int*)(dst + (size_t)row * rowlen);
  for (int i = threadIdx.x; i < nvec; i += 256) {
    float4 v = srow[i];
    drow[i] = clamp_q(v.x, inv) | (clamp_q(v.y, inv) << 8) |
              (clamp_q(v.z, inv) << 16) | (clamp_q(v.w, inv) << 24);
  }
}

#define BM 128
#define BN 128
#define BK 64
#define LDS_STRIDE 80

__device__ __forceinline__ void async_copy16(const int8_t* g, int8_t* l) {
  uint32_t lds_off = (uint32_t)(uintptr_t)l;
  uint64_t gaddr   = (uint64_t)(uintptr_t)g;
  asm volatile("global_load_async_to_lds_b128 %0, %1, off"
               :: "v"(lds_off), "v"(gaddr) : "memory");
}

__device__ __forceinline__ void wait_asynccnt0() {
  asm volatile("s_wait_asynccnt 0x0" ::: "memory");
}

__global__ void __launch_bounds__(256)
gemm_w8a8_wmma(const int8_t* __restrict__ qx, const int8_t* __restrict__ qw,
               const float* __restrict__ iscale, const float* __restrict__ wscale,
               const float* __restrict__ bias, float* __restrict__ out,
               int M, int N, int K) {
  __shared__ alignas(16) int8_t As[2][BM * LDS_STRIDE];
  __shared__ alignas(16) int8_t Bs[2][BN * LDS_STRIDE];

  const int tid  = threadIdx.x;
  const int lane = tid & 31;
  const int wave = tid >> 5;
  const int wm   = wave & 1;
  const int wn   = wave >> 1;
  const int lr   = lane & 15;
  const int lh   = lane >> 4;

  const int m0 = blockIdx.y * BM;
  const int n0 = blockIdx.x * BN;

  const int c0 = tid, c1 = tid + 256;
  const int rr0 = c0 >> 2, oo0 = (c0 & 3) << 4;
  const int rr1 = c1 >> 2, oo1 = (c1 & 3) << 4;

  const int8_t* gA = qx + (size_t)m0 * K;
  const int8_t* gB = qw + (size_t)n0 * K;

  vi8 acc[4][2];
#pragma unroll
  for (int mi = 0; mi < 4; ++mi)
#pragma unroll
    for (int ni = 0; ni < 2; ++ni) acc[mi][ni] = (vi8)0;

  async_copy16(gA + (size_t)rr0 * K + oo0, &As[0][rr0 * LDS_STRIDE + oo0]);
  async_copy16(gA + (size_t)rr1 * K + oo1, &As[0][rr1 * LDS_STRIDE + oo1]);
  async_copy16(gB + (size_t)rr0 * K + oo0, &Bs[0][rr0 * LDS_STRIDE + oo0]);
  async_copy16(gB + (size_t)rr1 * K + oo1, &Bs[0][rr1 * LDS_STRIDE + oo1]);
  wait_asynccnt0();
  __syncthreads();

  const int nk = K / BK;
  for (int kt = 0; kt < nk; ++kt) {
    const int cur = kt & 1;

    if (kt + 1 < nk) {
      const int nxt = cur ^ 1;
      const int kb = (kt + 1) * BK;
      async_copy16(gA + (size_t)rr0 * K + kb + oo0, &As[nxt][rr0 * LDS_STRIDE + oo0]);
      async_copy16(gA + (size_t)rr1 * K + kb + oo1, &As[nxt][rr1 * LDS_STRIDE + oo1]);
      async_copy16(gB + (size_t)rr0 * K + kb + oo0, &Bs[nxt][rr0 * LDS_STRIDE + oo0]);
      async_copy16(gB + (size_t)rr1 * K + kb + oo1, &Bs[nxt][rr1 * LDS_STRIDE + oo1]);
    }

    vi8 bfrag[2];
#pragma unroll
    for (int ni = 0; ni < 2; ++ni) {
      const int8_t* bb = &Bs[cur][(wn * 32 + ni * 16 + lr) * LDS_STRIDE + lh * 16];
      int4 q0 = *(const int4*)(bb);
      int4 q1 = *(const int4*)(bb + 32);
      vi8 b;
      b[0] = q0.x; b[1] = q0.y; b[2] = q0.z; b[3] = q0.w;
      b[4] = q1.x; b[5] = q1.y; b[6] = q1.z; b[7] = q1.w;
      bfrag[ni] = b;
    }

#pragma unroll
    for (int mi = 0; mi < 4; ++mi) {
      const int8_t* ab = &As[cur][(wm * 64 + mi * 16 + lr) * LDS_STRIDE + lh * 16];
      int2 p0 = *(const int2*)(ab);
      int2 p1 = *(const int2*)(ab + 8);
      int2 p2 = *(const int2*)(ab + 32);
      int2 p3 = *(const int2*)(ab + 40);
      vi8 a;
      a[0] = p0.x; a[1] = p0.y; a[2] = p1.x; a[3] = p1.y;
      a[4] = p2.x; a[5] = p2.y; a[6] = p3.x; a[7] = p3.y;
      acc[mi][0] = __builtin_amdgcn_wmma_i32_16x16x64_iu8(true, a, true, bfrag[0],
                                                          acc[mi][0], false, false);
      acc[mi][1] = __builtin_amdgcn_wmma_i32_16x16x64_iu8(true, a, true, bfrag[1],
                                                          acc[mi][1], false, false);
    }

    wait_asynccnt0();
    __syncthreads();
  }

#pragma unroll
  for (int mi = 0; mi < 4; ++mi) {
    const int rbase = m0 + wm * 64 + mi * 16 + lh * 8;
    float is[8];
#pragma unroll
    for (int e = 0; e < 8; ++e) is[e] = iscale[rbase + e];
#pragma unroll
    for (int ni = 0; ni < 2; ++ni) {
      const int col = n0 + wn * 32 + ni * 16 + lr;
      const float wsc = wscale[col];
      const float bv  = bias[col];
#pragma unroll
      for (int e = 0; e < 8; ++e) {
        out[(size_t)(rbase + e) * N + col] =
            (float)acc[mi][ni][e] * is[e] * wsc + bv;
      }
    }
  }
}

extern "C" void kernel_launch(void* const* d_in, const int* in_sizes, int n_in,
                              void* d_out, int out_size, void* d_ws, size_t ws_size,
                              hipStream_t stream) {
  const float* x    = (const float*)d_in[0];
  const float* w    = (const float*)d_in[1];
  const float* bias = (const float*)d_in[2];
  float* out = (float*)d_out;

  const int N = in_sizes[2];
  const int K = in_sizes[1] / N;
  const int M = in_sizes[0] / K;

  char* ws = (char*)d_ws;
  int8_t* qx = (int8_t*)ws;
  int8_t* qw = (int8_t*)(ws + (size_t)M * K);
  float* isc = (float*)(ws + (size_t)M * K + (size_t)N * K);
  float* wsc = isc + M;

  quant_rows_i8<<<N, 256, 0, stream>>>(w, qw, wsc, K, 0.0f);
  quant_rows_i8<<<M, 256, 0, stream>>>(x, qx, isc, K, 1e-5f);
  gemm_w8a8_wmma<<<dim3(N / BN, M / BM), 256, 0, stream>>>(qx, qw, isc, wsc, bias,
                                                           out, M, N, K);
}
